// Classifier_26723286516422
// MI455X (gfx1250) — matrix operand order repaired and checked by reading, not compiled, not run on hardware
//
#include <hip/hip_runtime.h>
#include <hip/hip_bf16.h>

typedef _Float16 v16h __attribute__((ext_vector_type(16)));
typedef _Float16 v8h  __attribute__((ext_vector_type(8)));
typedef float    v8f  __attribute__((ext_vector_type(8)));

#define LEAK 0.5f

// ---------------------------------------------------------------------------
// Prep: binarize + reorder conv2 weights into K=(ky,kx,ci) GEMM panel, f16.
// wA2[co][k], k = (ky*3+kx)*128 + ci,  co in [0,64), k in [0,1152)
// ---------------------------------------------------------------------------
__global__ void prep_w2_kernel(const float* __restrict__ w2, _Float16* __restrict__ wA2) {
    int idx = blockIdx.x * blockDim.x + threadIdx.x;
    if (idx >= 64 * 1152) return;
    int co = idx / 1152;
    int k  = idx % 1152;
    int ci = k & 127;
    int p  = k >> 7;            // 0..8
    int ky = p / 3, kx = p % 3;
    float w = w2[((co * 128 + ci) * 3 + ky) * 3 + kx];
    wA2[idx] = (_Float16)((w >= 0.f) ? 1.f : -1.f);
}

// wA3[co][k], k = (ky*3+kx)*64 + ci, co in [0,32), k in [0,576)
__global__ void prep_w3_kernel(const float* __restrict__ w3, _Float16* __restrict__ wA3) {
    int idx = blockIdx.x * blockDim.x + threadIdx.x;
    if (idx >= 32 * 576) return;
    int co = idx / 576;
    int k  = idx % 576;
    int ci = k & 63;
    int p  = k >> 6;            // 0..8
    int ky = p / 3, kx = p % 3;
    float w = w3[((co * 64 + ci) * 3 + ky) * 3 + kx];
    wA3[idx] = (_Float16)((w >= 0.f) ? 1.f : -1.f);
}

// ---------------------------------------------------------------------------
// Layer 1: conv(1->128, 3x3) + bias + leaky + maxpool2, fused.
// Output a1: NHWC f16 [32][111][111][128].
// Block = 128 threads (one per out channel), one pooled pixel per block.
// ---------------------------------------------------------------------------
__global__ void __launch_bounds__(128) conv1_pool_kernel(
    const float* __restrict__ x, const float* __restrict__ w1,
    const float* __restrict__ b1, _Float16* __restrict__ a1) {
    __shared__ float patch[16];                 // 4x4 input window
    const int px = blockIdx.x;                  // pooled col 0..110
    const int py = blockIdx.y;                  // pooled row 0..110
    const int b  = blockIdx.z;
    const int c  = threadIdx.x;                 // out channel 0..127
    if (c < 16) {
        int r = c >> 2, q = c & 3;
        patch[c] = x[((size_t)b * 224 + (2 * py + r)) * 224 + (2 * px + q)];
    }
    __syncthreads();
    float ws[9];
#pragma unroll
    for (int i = 0; i < 9; ++i) ws[i] = (w1[c * 9 + i] >= 0.f) ? 1.f : -1.f;
    const float bias = b1[c];
    float m = -3.0e38f;
#pragma unroll
    for (int pyy = 0; pyy < 2; ++pyy)
#pragma unroll
        for (int pxx = 0; pxx < 2; ++pxx) {
            float s = bias;
#pragma unroll
            for (int ky = 0; ky < 3; ++ky)
#pragma unroll
                for (int kx = 0; kx < 3; ++kx)
                    s += ws[ky * 3 + kx] * patch[(pyy + ky) * 4 + (pxx + kx)];
            s = (s >= 0.f) ? s : LEAK * s;
            m = fmaxf(m, s);
        }
    a1[(((size_t)b * 111 + py) * 111 + px) * 128 + c] = (_Float16)m;
}

// ---------------------------------------------------------------------------
// Layer 2 conv+leaky+maxpool fused, WMMA GEMM core.
// C[64 x (108*108 used)] = Wb[64 x 1152] * im2col(a1), pooled to [54,54].
// Block: 256 thr = 8 waves = 2 conv rows x 4 M-tiles of 16 out-channels.
// Each wave: 16 ch x 16 cols tile, K=1152 in 36 WMMA steps.
// B-fragment = 32 contiguous bytes/lane (NHWC + (ky,kx,ci) K-order).
// Bias folded into acc init; leaky before LDS; 2x2 max-reduce -> a2 f16 NHWC.
// ---------------------------------------------------------------------------
__global__ void __launch_bounds__(256) conv2_wmma_pool_kernel(
    const _Float16* __restrict__ a1, const _Float16* __restrict__ wA,
    const float* __restrict__ bias, _Float16* __restrict__ a2) {
    __shared__ float tile[2][64][16];         // [conv row][channel][col]
    const int b    = blockIdx.z;
    const int py   = blockIdx.y;              // pooled row 0..53
    const int ox0  = blockIdx.x * 16;         // conv col base 0,16,...,96
    const int wave = threadIdx.x >> 5;
    const int lane = threadIdx.x & 31;
    const int r    = wave >> 2;               // conv row within pooled pair
    const int co0  = (wave & 3) * 16;         // 0,16,32,48
    const int nl   = lane & 15;
    const int hi   = lane >> 4;               // 0/1 half-wave
    const int oy   = 2 * py + r;              // conv row 0..107
    const int ocol = min(ox0 + nl, 107);      // clamp edge lanes (dup column)

    // 16x16 f32 C layout: lanes 0-15 -> M=v, lanes 16-31 -> M=v+8
    v8f acc;
#pragma unroll
    for (int v = 0; v < 8; ++v) acc[v] = bias[co0 + hi * 8 + v];

    const _Float16* wrow = wA + (size_t)(co0 + nl) * 1152;
    const int aoff = hi * 16;                 // A layout: lo lanes K 0..15, hi lanes K 16..31

    for (int kc = 0; kc < 36; ++kc) {
        const int k0  = kc * 32;
        const int p   = kc >> 2;              // 0..8  (ky*3+kx)
        const int ci0 = (kc & 3) * 32;        // 0,32,64,96
        const int ky  = p / 3, kx = p % 3;

        // A fragment: one 32B contiguous chunk per lane
        v16h afrag = *(const v16h*)(wrow + k0 + aoff);

        // B fragment: 16 contiguous channels at pixel (oy+ky, ocol+kx)
        const _Float16* bp = a1 + ((((size_t)b * 111 + (oy + ky)) * 111 + (ocol + kx)) << 7)
                                + ci0 + hi * 16;
        v16h bfrag = *(const v16h*)bp;
        __builtin_prefetch(bp + 128, 0, 3);   // next-pixel channel block

        acc = __builtin_amdgcn_wmma_f32_16x16x32_f16(
                  false, afrag, false, bfrag, (short)0, acc, false, false);
    }

    // leaky + stash tile in LDS
#pragma unroll
    for (int v = 0; v < 8; ++v) {
        float val = acc[v];
        val = (val >= 0.f) ? val : LEAK * val;
        tile[r][co0 + hi * 8 + v][nl] = val;
    }
    __syncthreads();

    // 2x2 max-pool: 8 pooled cols x 64 channels = 512 outputs
    for (int e = threadIdx.x; e < 512; e += 256) {
        int c  = e & 63;
        int pp = e >> 6;                      // 0..7 pooled col within tile
        int px = (ox0 >> 1) + pp;             // global pooled col
        if (px < 54) {
            float m = fmaxf(fmaxf(tile[0][c][2 * pp], tile[0][c][2 * pp + 1]),
                            fmaxf(tile[1][c][2 * pp], tile[1][c][2 * pp + 1]));
            a2[(((size_t)b * 54 + py) * 54 + px) * 64 + c] = (_Float16)m;
        }
    }
}

// ---------------------------------------------------------------------------
// Layer 3 conv+leaky+maxpool fused WMMA: Wb[32 x 576] * im2col(a2) -> a3 f32.
// Block: 128 thr = 4 waves = 2 conv rows x 2 M-tiles.
// ---------------------------------------------------------------------------
__global__ void __launch_bounds__(128) conv3_wmma_pool_kernel(
    const _Float16* __restrict__ a2, const _Float16* __restrict__ wA,
    const float* __restrict__ bias, float* __restrict__ a3) {
    __shared__ float tile[2][32][16];
    const int b    = blockIdx.z;
    const int py   = blockIdx.y;              // pooled row 0..25
    const int ox0  = blockIdx.x * 16;         // 0,16,32,48
    const int wave = threadIdx.x >> 5;
    const int lane = threadIdx.x & 31;
    const int r    = wave >> 1;               // conv row within pooled pair
    const int co0  = (wave & 1) * 16;         // 0,16
    const int nl   = lane & 15;
    const int hi   = lane >> 4;
    const int oy   = 2 * py + r;              // conv row 0..51
    const int ocol = min(ox0 + nl, 51);

    v8f acc;
#pragma unroll
    for (int v = 0; v < 8; ++v) acc[v] = bias[co0 + hi * 8 + v];

    const _Float16* wrow = wA + (size_t)(co0 + nl) * 576;
    const int aoff = hi * 16;

    for (int kc = 0; kc < 18; ++kc) {
        const int k0  = kc * 32;
        const int p   = kc >> 1;              // 0..8
        const int ci0 = (kc & 1) * 32;        // 0,32
        const int ky  = p / 3, kx = p % 3;

        v16h afrag = *(const v16h*)(wrow + k0 + aoff);

        const _Float16* bp = a2 + ((((size_t)b * 54 + (oy + ky)) * 54 + (ocol + kx)) << 6)
                                + ci0 + hi * 16;
        v16h bfrag = *(const v16h*)bp;
        __builtin_prefetch(bp + 64, 0, 3);

        acc = __builtin_amdgcn_wmma_f32_16x16x32_f16(
                  false, afrag, false, bfrag, (short)0, acc, false, false);
    }

#pragma unroll
    for (int v = 0; v < 8; ++v) {
        float val = acc[v];
        val = (val >= 0.f) ? val : LEAK * val;
        tile[r][co0 + hi * 8 + v][nl] = val;
    }
    __syncthreads();

    // 2x2 max-pool: 8 pooled cols x 32 channels = 256 outputs
    for (int e = threadIdx.x; e < 256; e += 128) {
        int c  = e & 31;
        int pp = e >> 5;                      // 0..7
        int px = (ox0 >> 1) + pp;
        if (px < 26) {
            float m = fmaxf(fmaxf(tile[0][c][2 * pp], tile[0][c][2 * pp + 1]),
                            fmaxf(tile[1][c][2 * pp], tile[1][c][2 * pp + 1]));
            a3[(((size_t)b * 26 + py) * 26 + px) * 32 + c] = m;
        }
    }
}

// ---------------------------------------------------------------------------
// Layer 4: conv(32->8, 3x2) + bias, direct. out[b*4800 + c*600 + y*25 + x].
// ---------------------------------------------------------------------------
__global__ void conv4_kernel(const float* __restrict__ a3, const float* __restrict__ w4,
                             const float* __restrict__ b4, float* __restrict__ out) {
    int idx = blockIdx.x * blockDim.x + threadIdx.x;
    if (idx >= 32 * 8 * 24 * 25) return;
    int x = idx % 25;
    int y = (idx / 25) % 24;
    int c = (idx / (25 * 24)) % 8;
    int b = idx / (25 * 24 * 8);
    float s = b4[c];
    for (int ci = 0; ci < 32; ++ci) {
#pragma unroll
        for (int ky = 0; ky < 3; ++ky)
#pragma unroll
            for (int kx = 0; kx < 2; ++kx) {
                float w  = w4[((c * 32 + ci) * 3 + ky) * 2 + kx];
                float wv = (w >= 0.f) ? 1.f : -1.f;
                s += wv * a3[(((size_t)b * 26 + (y + ky)) * 26 + (x + kx)) * 32 + ci];
            }
    }
    out[idx] = s;
}

// ---------------------------------------------------------------------------
extern "C" void kernel_launch(void* const* d_in, const int* in_sizes, int n_in,
                              void* d_out, int out_size, void* d_ws, size_t ws_size,
                              hipStream_t stream) {
    const float* x  = (const float*)d_in[0];
    const float* w1 = (const float*)d_in[1];
    const float* b1 = (const float*)d_in[2];
    const float* w2 = (const float*)d_in[3];
    const float* b2 = (const float*)d_in[4];
    const float* w3 = (const float*)d_in[5];
    const float* b3 = (const float*)d_in[6];
    const float* w4 = (const float*)d_in[7];
    const float* b4 = (const float*)d_in[8];
    float* out = (float*)d_out;

    // workspace layout (all offsets 256B-aligned)
    char* ws = (char*)d_ws;
    _Float16* wA2 = (_Float16*)(ws + 0);           //     147,456 B
    _Float16* wA3 = (_Float16*)(ws + 147456);      //      36,864 B
    _Float16* a1  = (_Float16*)(ws + 184320);      // 100,933,632 B  [32][111][111][128]
    _Float16* a2  = (_Float16*)(ws + 101117952);   //  11,943,936 B  [32][54][54][64]
    float*    a3  = (float*)   (ws + 113061888);   //   2,768,896 B  [32][26][26][32]

    // weight prep
    prep_w2_kernel<<<(64 * 1152 + 255) / 256, 256, 0, stream>>>(w2, wA2);
    prep_w3_kernel<<<(32 * 576 + 255) / 256, 256, 0, stream>>>(w3, wA3);

    // layer 1 fused conv+leaky+pool
    conv1_pool_kernel<<<dim3(111, 111, 32), 128, 0, stream>>>(x, w1, b1, a1);

    // layer 2 fused WMMA conv + leaky + pool
    conv2_wmma_pool_kernel<<<dim3(7, 54, 32), 256, 0, stream>>>(a1, wA2, b2, a2);

    // layer 3 fused WMMA conv + leaky + pool
    conv3_wmma_pool_kernel<<<dim3(4, 26, 32), 128, 0, stream>>>(a2, wA3, b3, a3);

    // layer 4 direct conv -> flattened output
    conv4_kernel<<<(32 * 8 * 24 * 25 + 255) / 256, 256, 0, stream>>>(a3, w4, b4, out);
}
